// DynamicGraphNetwork_65661460021917
// MI455X (gfx1250) — matrix operand order repaired and checked by reading, not compiled, not run on hardware
//
#include <hip/hip_runtime.h>

// ---------------------------------------------------------------------------
// Problem constants (from the reference)
// ---------------------------------------------------------------------------
#define NN 50000
#define EE 800000
#define BB 8
#define HQ_ 32
#define LN_EPS 1e-5f

typedef __attribute__((ext_vector_type(16))) __bf16 v16bf;
typedef __attribute__((ext_vector_type(8)))  float  v8f;
typedef __attribute__((ext_vector_type(4)))  unsigned int v4u;

union FragU { v4u u[2]; v16bf v; };

__device__ __forceinline__ v8f zero8() {
  v8f v;
#pragma unroll
  for (int i = 0; i < 8; ++i) v[i] = 0.0f;
  return v;
}

__device__ __forceinline__ float silu_f(float x) {
  return x / (1.0f + __expf(-x));
}

__device__ __forceinline__ v8f wmma_bf16(v16bf a, v16bf b, v8f c) {
  return __builtin_amdgcn_wmma_f32_16x16x32_bf16(false, a, false, b,
                                                 (short)0, c, false, false);
}

// A fragment: 16x32 bf16 tile from row-major LDS [16][lda] at K offset kbase.
// Per ISA: lanes 0-15 (M=lane) hold K = kbase+{0..7, 16..23};
//          lanes 16-31 hold K = kbase+{8..15, 24..31}.
__device__ __forceinline__ v16bf frag_a(const __bf16* sA, int lda, int kbase, int lane) {
  const __bf16* p = sA + (lane & 15) * lda + kbase + ((lane & 16) ? 8 : 0);
  FragU f;
  f.u[0] = *(const v4u*)(p);
  f.u[1] = *(const v4u*)(p + 16);
  return f.v;
}

// B fragment: 32x16 bf16 tile from W^T [Nout][ldk] (transposed, K contiguous).
// Lane holds column N = nbase + (lane&15); lanes 0-15: K = kbase+{0..7, 16..23},
// lanes 16-31: K = kbase+{8..15, 24..31} (same order as frag_a).
__device__ __forceinline__ v16bf frag_b(const __bf16* wT, int ldk, int nbase,
                                        int kbase, int lane) {
  const __bf16* p = wT + (size_t)(nbase + (lane & 15)) * ldk + kbase + ((lane & 16) ? 8 : 0);
  FragU f;
  f.u[0] = *(const v4u*)(p);
  f.u[1] = *(const v4u*)(p + 16);
  return f.v;
}

// 2 M-tiles (32 rows) x 4 N-tiles (64 cols) per wave; B frags reused across M.
template <int KPAD>
__device__ __forceinline__ void gemm_m2n4(const __bf16* sA, const __bf16* wT,
                                          int nbase, int lane, v8f acc[2][4]) {
  for (int kb = 0; kb < KPAD; kb += 32) {
    v16bf a0 = frag_a(sA, KPAD, kb, lane);
    v16bf a1 = frag_a(sA + 16 * KPAD, KPAD, kb, lane);
#pragma unroll
    for (int j = 0; j < 4; ++j) {
      v16bf b = frag_b(wT, KPAD, nbase + j * 16, kb, lane);
      acc[0][j] = wmma_bf16(a0, b, acc[0][j]);
      acc[1][j] = wmma_bf16(a1, b, acc[1][j]);
    }
  }
}

// 2 M-tiles x 1 N-tile per wave
template <int KPAD>
__device__ __forceinline__ void gemm_m2n1(const __bf16* sA, const __bf16* wT,
                                          int nbase, int lane, v8f acc[2]) {
  for (int kb = 0; kb < KPAD; kb += 32) {
    v16bf a0 = frag_a(sA, KPAD, kb, lane);
    v16bf a1 = frag_a(sA + 16 * KPAD, KPAD, kb, lane);
    v16bf b = frag_b(wT, KPAD, nbase, kb, lane);
    acc[0] = wmma_bf16(a0, b, acc[0]);
    acc[1] = wmma_bf16(a1, b, acc[1]);
  }
}

// D layout: element r -> row = rowbase + r + (lane&16 ? 8 : 0),
//           col = nbase + (lane&15)
__device__ __forceinline__ void silu_store_bf16(__bf16* s, int ld, int nbase,
                                                int rowbase, v8f c,
                                                const float* bias, int lane) {
  int col = nbase + (lane & 15);
  float bv = bias[col];
  int rb = rowbase + ((lane & 16) ? 8 : 0);
#pragma unroll
  for (int r = 0; r < 8; ++r) {
    float v = silu_f(c[r] + bv);
    s[(rb + r) * ld + col] = (__bf16)v;
  }
}

// ---------------------------------------------------------------------------
// Utility kernels
// ---------------------------------------------------------------------------
__global__ void zero_f(float* p, int n) {
  int i = blockIdx.x * blockDim.x + threadIdx.x;
  if (i < n) p[i] = 0.0f;
}

// src: fp32 [K][Nout] (fan_in x fan_out). dst: bf16 [Nout][Kpad] transposed, zero-padded.
__global__ void convert_wT(const float* __restrict__ src, __bf16* __restrict__ dst,
                           int K, int Nout, int Kpad) {
  int idx = blockIdx.x * blockDim.x + threadIdx.x;
  if (idx >= Nout * Kpad) return;
  int n = idx / Kpad, k = idx - n * Kpad;
  float v = (k < K) ? src[(size_t)k * Nout + n] : 0.0f;
  dst[idx] = (__bf16)v;
}

// ---------------------------------------------------------------------------
// Time embedding + time MLP  (B=8, HQ=32, hidden 256) -> tmlp [8][32]
// ---------------------------------------------------------------------------
__global__ void __launch_bounds__(256)
time_kernel(const float* __restrict__ t,
            const float* __restrict__ tW1, const float* __restrict__ tb1,
            const float* __restrict__ tW2, const float* __restrict__ tb2,
            float* __restrict__ tmlp) {
  __shared__ float sEmb[BB * HQ_];
  __shared__ float sH1[BB * 256];
  int tid = threadIdx.x;
  if (tid < BB * HQ_) {
    int b = tid >> 5, j = tid & 31;
    float tv = t[b];
    int jj = (j < 16) ? j : (j - 16);
    float fr = __expf(-9.210340372f * (float)jj / 16.0f);  // log(10000)
    sEmb[tid] = (j < 16) ? __cosf(tv * fr) : __sinf(tv * fr);
  }
  __syncthreads();
  for (int i = tid; i < BB * 256; i += 256) {
    int b = i >> 8, m = i & 255;
    float s = tb1[m];
    for (int j = 0; j < HQ_; ++j) s += sEmb[b * HQ_ + j] * tW1[j * 256 + m];
    sH1[i] = silu_f(s);
  }
  __syncthreads();
  if (tid < BB * HQ_) {
    int b = tid >> 5, q = tid & 31;
    float s = tb2[q];
    for (int m = 0; m < 256; ++m) s += sH1[b * 256 + m] * tW2[m * HQ_ + q];
    tmlp[tid] = s;
  }
}

// ---------------------------------------------------------------------------
// Fused message kernel: gather -> message MLP (449->512->128, WMMA) ->
//   scatter-add aggr + pos MLP (128->512->3) + pos scatter + deg count
// Block = 256 threads (8 waves), 32 edges per block.
// LDS: sBuf reused as (A 32x512) -> (h1 32x512) -> (h2 32x512)
// ---------------------------------------------------------------------------
__global__ void __launch_bounds__(256)
msg_kernel(const float* __restrict__ x_nodes, const float* __restrict__ edge_attr,
           const float* __restrict__ dist, const float* __restrict__ frame_emb,
           const float* __restrict__ pos, const int* __restrict__ batch,
           const int* __restrict__ ei, const float* __restrict__ tmlp,
           const __bf16* __restrict__ mW1T, const float* __restrict__ mb1,
           const __bf16* __restrict__ mW2T, const float* __restrict__ mb2,
           const __bf16* __restrict__ pW1T, const float* __restrict__ pb1,
           const float* __restrict__ pW2, const float* __restrict__ pb2,
           float* __restrict__ aggr, float* __restrict__ pos_sum,
           float* __restrict__ deg, int E_) {
  __shared__ alignas(16) __bf16 sBuf[32 * 512];   // 32KB, multi-purpose
  __shared__ alignas(16) float  sMsgF[32 * 128];  // messages fp32 (16KB)
  __shared__ alignas(16) __bf16 sMsgB[32 * 128];  // messages bf16 (8KB)
  __shared__ int sSrc[32], sDst[32];

  const int e0 = blockIdx.x * 32;
  const int tid = threadIdx.x;
  const int lane = tid & 31, wv = tid >> 5;

  if (tid < 32) {
    int e = e0 + tid;
    sSrc[tid] = (e < E_) ? ei[e] : 0;
    sDst[tid] = (e < E_) ? ei[E_ + e] : 0;
  }
  __syncthreads();

  // Stage A: concat [x_src(128) | x_dst(128) | edge_attr(128) | dist(1) |
  //                  t_emb(32) | frame(32) | zeros(63)]
  for (int i = tid; i < 32 * 512; i += 256) {
    int r = i >> 9, k = i & 511;
    int e = e0 + r;
    float v = 0.0f;
    if (e < E_) {
      int s = sSrc[r], d = sDst[r];
      if (k < 128)       v = x_nodes[(size_t)s * 128 + k];
      else if (k < 256)  v = x_nodes[(size_t)d * 128 + (k - 128)];
      else if (k < 384)  v = edge_attr[(size_t)e * 128 + (k - 256)];
      else if (k == 384) v = dist[e];
      else if (k < 417)  v = tmlp[batch[s] * HQ_ + (k - 385)];
      else if (k < 449)  v = frame_emb[(size_t)s * HQ_ + (k - 417)];
    }
    sBuf[i] = (__bf16)v;
  }
  __syncthreads();

  // Layer 1: 512(pad of 449) -> 512, SiLU; hidden overwrites sBuf
  {
    v8f acc[2][4] = {{zero8(), zero8(), zero8(), zero8()},
                     {zero8(), zero8(), zero8(), zero8()}};
    gemm_m2n4<512>(sBuf, mW1T, wv * 64, lane, acc);
    __syncthreads();  // all reads of sBuf (A) done
#pragma unroll
    for (int m = 0; m < 2; ++m)
#pragma unroll
      for (int j = 0; j < 4; ++j)
        silu_store_bf16(sBuf, 512, wv * 64 + j * 16, m * 16, acc[m][j], mb1, lane);
  }
  __syncthreads();

  // Layer 2: 512 -> 128 messages, SiLU
  {
    v8f acc[2] = {zero8(), zero8()};
    gemm_m2n1<512>(sBuf, mW2T, wv * 16, lane, acc);
    int col = wv * 16 + (lane & 15);
    float bv = mb2[col];
#pragma unroll
    for (int m = 0; m < 2; ++m) {
      int rb = m * 16 + ((lane & 16) ? 8 : 0);
#pragma unroll
      for (int r = 0; r < 8; ++r) {
        float v = silu_f(acc[m][r] + bv);
        sMsgF[(rb + r) * 128 + col] = v;
        sMsgB[(rb + r) * 128 + col] = (__bf16)v;
      }
    }
  }
  __syncthreads();  // messages final; sBuf reads done

  // Scatter-add aggr[dst] += messages
  for (int i = tid; i < 32 * 128; i += 256) {
    int r = i >> 7, c = i & 127;
    int e = e0 + r;
    if (e < E_) atomicAdd(&aggr[(size_t)sDst[r] * 128 + c], sMsgF[i]);
  }

  // pos MLP layer 1: messages(32x128) -> h2(32x512), SiLU, into sBuf
  {
    v8f acc[2][4] = {{zero8(), zero8(), zero8(), zero8()},
                     {zero8(), zero8(), zero8(), zero8()}};
    gemm_m2n4<128>(sMsgB, pW1T, wv * 64, lane, acc);
#pragma unroll
    for (int m = 0; m < 2; ++m)
#pragma unroll
      for (int j = 0; j < 4; ++j)
        silu_store_bf16(sBuf, 512, wv * 64 + j * 16, m * 16, acc[m][j], pb1, lane);
  }
  __syncthreads();

  // pos MLP layer 2: 512 -> 3, multiply by (pos[dst]-pos[src]), scatter
  if (tid < 96) {
    int r = tid / 3, d = tid - 3 * (tid / 3);
    int e = e0 + r;
    float sum = pb2[d];
    for (int k = 0; k < 512; ++k) sum += (float)sBuf[r * 512 + k] * pW2[k * 3 + d];
    if (e < E_) {
      int s = sSrc[r], dd = sDst[r];
      float delta = pos[dd * 3 + d] - pos[s * 3 + d];
      atomicAdd(&pos_sum[dd * 3 + d], sum * delta);
    }
  }
  if (tid < 32) {
    int e = e0 + tid;
    if (e < E_) atomicAdd(&deg[sDst[tid]], 1.0f);
  }
}

// ---------------------------------------------------------------------------
// Node kernel: concat(x, aggr, t_emb) 288 -> 512 -> 128 + residual LayerNorm
// 32 nodes per block.
// ---------------------------------------------------------------------------
__global__ void __launch_bounds__(256)
node_kernel(const float* __restrict__ x_nodes, const float* __restrict__ aggr,
            const int* __restrict__ batch, const float* __restrict__ tmlp,
            const __bf16* __restrict__ nW1T, const float* __restrict__ nb1,
            const __bf16* __restrict__ nW2T, const float* __restrict__ nb2,
            const float* __restrict__ g, const float* __restrict__ bsh,
            float* __restrict__ x_new, int N_) {
  __shared__ alignas(16) __bf16 sBuf[32 * 512];   // A (32x288) then hidden (32x512)
  __shared__ float sOut[32 * 128];
  __shared__ float sMu[32], sRs[32];
  const int n0 = blockIdx.x * 32;
  const int tid = threadIdx.x, lane = tid & 31, wv = tid >> 5;

  for (int i = tid; i < 32 * 288; i += 256) {
    int r = i / 288, k = i - r * 288;
    int n = n0 + r;
    float v = 0.0f;
    if (n < N_) {
      if (k < 128)      v = x_nodes[(size_t)n * 128 + k];
      else if (k < 256) v = aggr[(size_t)n * 128 + (k - 128)];
      else              v = tmlp[batch[n] * HQ_ + (k - 256)];
    }
    sBuf[i] = (__bf16)v;
  }
  __syncthreads();

  {
    v8f acc[2][4] = {{zero8(), zero8(), zero8(), zero8()},
                     {zero8(), zero8(), zero8(), zero8()}};
    gemm_m2n4<288>(sBuf, nW1T, wv * 64, lane, acc);
    __syncthreads();
#pragma unroll
    for (int m = 0; m < 2; ++m)
#pragma unroll
      for (int j = 0; j < 4; ++j)
        silu_store_bf16(sBuf, 512, wv * 64 + j * 16, m * 16, acc[m][j], nb1, lane);
  }
  __syncthreads();

  {
    v8f acc[2] = {zero8(), zero8()};
    gemm_m2n1<512>(sBuf, nW2T, wv * 16, lane, acc);
    int col = wv * 16 + (lane & 15);
    float bv = nb2[col];
#pragma unroll
    for (int m = 0; m < 2; ++m) {
      int rb = m * 16 + ((lane & 16) ? 8 : 0);
#pragma unroll
      for (int r = 0; r < 8; ++r) {
        int n = n0 + rb + r;
        float res = (n < N_) ? x_nodes[(size_t)n * 128 + col] : 0.0f;
        sOut[(rb + r) * 128 + col] = acc[m][r] + bv + res;
      }
    }
  }
  __syncthreads();

  if (tid < 32) {
    int r = tid;
    float m = 0.0f;
    for (int c = 0; c < 128; ++c) m += sOut[r * 128 + c];
    m *= (1.0f / 128.0f);
    float var = 0.0f;
    for (int c = 0; c < 128; ++c) { float d = sOut[r * 128 + c] - m; var += d * d; }
    var *= (1.0f / 128.0f);
    sMu[r] = m;
    sRs[r] = rsqrtf(var + LN_EPS);
  }
  __syncthreads();

  for (int i = tid; i < 32 * 128; i += 256) {
    int r = i >> 7, c = i & 127;
    int n = n0 + r;
    if (n < N_) x_new[(size_t)n * 128 + c] = (sOut[i] - sMu[r]) * sRs[r] * g[c] + bsh[c];
  }
}

// ---------------------------------------------------------------------------
// Edge kernel: concat(x_new[src], x_new[dst], dist, t_emb) 289(pad 320)
//   -> 512 -> 128 + residual LayerNorm. 32 edges per block.
// ---------------------------------------------------------------------------
__global__ void __launch_bounds__(256)
edge_kernel(const float* __restrict__ xn, const float* __restrict__ edge_attr,
            const float* __restrict__ dist, const int* __restrict__ batch,
            const int* __restrict__ ei, const float* __restrict__ tmlp,
            const __bf16* __restrict__ eW1T, const float* __restrict__ eb1,
            const __bf16* __restrict__ eW2T, const float* __restrict__ eb2,
            const float* __restrict__ g, const float* __restrict__ bsh,
            float* __restrict__ eout, int E_) {
  __shared__ alignas(16) __bf16 sBuf[32 * 512];   // A (32x320) then hidden (32x512)
  __shared__ float sOut[32 * 128];
  __shared__ float sMu[32], sRs[32];
  __shared__ int sSrc[32], sDst[32];
  const int e0 = blockIdx.x * 32;
  const int tid = threadIdx.x, lane = tid & 31, wv = tid >> 5;

  if (tid < 32) {
    int e = e0 + tid;
    sSrc[tid] = (e < E_) ? ei[e] : 0;
    sDst[tid] = (e < E_) ? ei[E_ + e] : 0;
  }
  __syncthreads();

  for (int i = tid; i < 32 * 320; i += 256) {
    int r = i / 320, k = i - r * 320;
    int e = e0 + r;
    float v = 0.0f;
    if (e < E_) {
      int s = sSrc[r], d = sDst[r];
      if (k < 128)       v = xn[(size_t)s * 128 + k];
      else if (k < 256)  v = xn[(size_t)d * 128 + (k - 128)];
      else if (k == 256) v = dist[e];
      else if (k < 289)  v = tmlp[batch[s] * HQ_ + (k - 257)];
    }
    sBuf[i] = (__bf16)v;
  }
  __syncthreads();

  {
    v8f acc[2][4] = {{zero8(), zero8(), zero8(), zero8()},
                     {zero8(), zero8(), zero8(), zero8()}};
    gemm_m2n4<320>(sBuf, eW1T, wv * 64, lane, acc);
    __syncthreads();
#pragma unroll
    for (int m = 0; m < 2; ++m)
#pragma unroll
      for (int j = 0; j < 4; ++j)
        silu_store_bf16(sBuf, 512, wv * 64 + j * 16, m * 16, acc[m][j], eb1, lane);
  }
  __syncthreads();

  {
    v8f acc[2] = {zero8(), zero8()};
    gemm_m2n1<512>(sBuf, eW2T, wv * 16, lane, acc);
    int col = wv * 16 + (lane & 15);
    float bv = eb2[col];
#pragma unroll
    for (int m = 0; m < 2; ++m) {
      int rb = m * 16 + ((lane & 16) ? 8 : 0);
#pragma unroll
      for (int r = 0; r < 8; ++r) {
        int e = e0 + rb + r;
        float res = (e < E_) ? edge_attr[(size_t)e * 128 + col] : 0.0f;
        sOut[(rb + r) * 128 + col] = acc[m][r] + bv + res;
      }
    }
  }
  __syncthreads();

  if (tid < 32) {
    int r = tid;
    float m = 0.0f;
    for (int c = 0; c < 128; ++c) m += sOut[r * 128 + c];
    m *= (1.0f / 128.0f);
    float var = 0.0f;
    for (int c = 0; c < 128; ++c) { float d = sOut[r * 128 + c] - m; var += d * d; }
    var *= (1.0f / 128.0f);
    sMu[r] = m;
    sRs[r] = rsqrtf(var + LN_EPS);
  }
  __syncthreads();

  for (int i = tid; i < 32 * 128; i += 256) {
    int r = i >> 7, c = i & 127;
    int e = e0 + r;
    if (e < E_) eout[(size_t)e * 128 + c] = (sOut[i] - sMu[r]) * sRs[r] * g[c] + bsh[c];
  }
}

// ---------------------------------------------------------------------------
// pos_v_t = pos_sum / clip(deg, 1)
// ---------------------------------------------------------------------------
__global__ void pos_final(const float* __restrict__ pos_sum, const float* __restrict__ deg,
                          float* __restrict__ out, int N_) {
  int i = blockIdx.x * blockDim.x + threadIdx.x;
  if (i >= N_ * 3) return;
  int n = i / 3;
  float d = deg[n];
  d = (d < 1.0f) ? 1.0f : d;
  out[i] = pos_sum[i] / d;
}

// ---------------------------------------------------------------------------
// Launcher
// ---------------------------------------------------------------------------
extern "C" void kernel_launch(void* const* d_in, const int* in_sizes, int n_in,
                              void* d_out, int out_size, void* d_ws, size_t ws_size,
                              hipStream_t stream) {
  const int N_ = NN, E_ = EE;
  const float* x_nodes   = (const float*)d_in[0];
  const float* edge_attr = (const float*)d_in[1];
  const float* dist      = (const float*)d_in[2];
  const float* frame_emb = (const float*)d_in[3];
  const float* pos       = (const float*)d_in[4];
  const float* t         = (const float*)d_in[5];
  const int*   batch     = (const int*)d_in[6];
  const int*   ei        = (const int*)d_in[7];
  const float* tW1 = (const float*)d_in[8];  const float* tb1 = (const float*)d_in[9];
  const float* tW2 = (const float*)d_in[10]; const float* tb2 = (const float*)d_in[11];
  const float* mW1 = (const float*)d_in[12]; const float* mb1 = (const float*)d_in[13];
  const float* mW2 = (const float*)d_in[14]; const float* mb2 = (const float*)d_in[15];
  const float* pW1 = (const float*)d_in[16]; const float* pb1 = (const float*)d_in[17];
  const float* pW2 = (const float*)d_in[18]; const float* pb2 = (const float*)d_in[19];
  const float* nW1 = (const float*)d_in[20]; const float* nb1 = (const float*)d_in[21];
  const float* nW2 = (const float*)d_in[22]; const float* nb2 = (const float*)d_in[23];
  const float* ln_n_g = (const float*)d_in[24]; const float* ln_n_b = (const float*)d_in[25];
  const float* eW1 = (const float*)d_in[26]; const float* eb1 = (const float*)d_in[27];
  const float* eW2 = (const float*)d_in[28]; const float* eb2 = (const float*)d_in[29];
  const float* ln_e_g = (const float*)d_in[30]; const float* ln_e_b = (const float*)d_in[31];

  // Output layout: pos_v_t [N*3] | x_new [N*128] | edge_new [E*128]
  float* out_pos  = (float*)d_out;
  float* out_x    = (float*)d_out + (size_t)N_ * 3;
  float* out_edge = (float*)d_out + (size_t)N_ * 3 + (size_t)N_ * 128;

  // Workspace carve-up
  char* ws = (char*)d_ws;
  size_t off = 0;
  auto take = [&](size_t bytes) -> char* {
    char* p = ws + off;
    off = (off + bytes + 255) & ~(size_t)255;
    return p;
  };
  float*  tmlp    = (float*)take((size_t)BB * HQ_ * 4);
  float*  aggr    = (float*)take((size_t)N_ * 128 * 4);
  float*  pos_sum = (float*)take((size_t)N_ * 3 * 4);
  float*  deg     = (float*)take((size_t)N_ * 4);
  __bf16* mW1T = (__bf16*)take((size_t)512 * 512 * 2);
  __bf16* mW2T = (__bf16*)take((size_t)128 * 512 * 2);
  __bf16* pW1T = (__bf16*)take((size_t)512 * 128 * 2);
  __bf16* nW1T = (__bf16*)take((size_t)512 * 288 * 2);
  __bf16* nW2T = (__bf16*)take((size_t)128 * 512 * 2);
  __bf16* eW1T = (__bf16*)take((size_t)512 * 320 * 2);
  __bf16* eW2T = (__bf16*)take((size_t)128 * 512 * 2);

  const int T = 256;
  // Zero accumulators
  zero_f<<<(N_ * 128 + T - 1) / T, T, 0, stream>>>(aggr, N_ * 128);
  zero_f<<<(N_ * 3 + T - 1) / T, T, 0, stream>>>(pos_sum, N_ * 3);
  zero_f<<<(N_ + T - 1) / T, T, 0, stream>>>(deg, N_);

  // Pre-convert weights to transposed, K-padded bf16
  convert_wT<<<(512 * 512 + T - 1) / T, T, 0, stream>>>(mW1, mW1T, 449, 512, 512);
  convert_wT<<<(128 * 512 + T - 1) / T, T, 0, stream>>>(mW2, mW2T, 512, 128, 512);
  convert_wT<<<(512 * 128 + T - 1) / T, T, 0, stream>>>(pW1, pW1T, 128, 512, 128);
  convert_wT<<<(512 * 288 + T - 1) / T, T, 0, stream>>>(nW1, nW1T, 288, 512, 288);
  convert_wT<<<(128 * 512 + T - 1) / T, T, 0, stream>>>(nW2, nW2T, 512, 128, 512);
  convert_wT<<<(512 * 320 + T - 1) / T, T, 0, stream>>>(eW1, eW1T, 289, 512, 320);
  convert_wT<<<(128 * 512 + T - 1) / T, T, 0, stream>>>(eW2, eW2T, 512, 128, 512);

  // Time MLP
  time_kernel<<<1, T, 0, stream>>>(t, tW1, tb1, tW2, tb2, tmlp);

  // Fused message + pos MLP + scatters (32 edges / block)
  msg_kernel<<<(E_ + 31) / 32, T, 0, stream>>>(
      x_nodes, edge_attr, dist, frame_emb, pos, batch, ei, tmlp,
      mW1T, mb1, mW2T, mb2, pW1T, pb1, pW2, pb2,
      aggr, pos_sum, deg, E_);

  // Node update (+residual LN) -> out_x (32 nodes / block)
  node_kernel<<<(N_ + 31) / 32, T, 0, stream>>>(
      x_nodes, aggr, batch, tmlp, nW1T, nb1, nW2T, nb2, ln_n_g, ln_n_b, out_x, N_);

  // Position scatter-mean finalize -> out_pos
  pos_final<<<(N_ * 3 + T - 1) / T, T, 0, stream>>>(pos_sum, deg, out_pos, N_);

  // Edge update (+residual LN) -> out_edge (32 edges / block)
  edge_kernel<<<(E_ + 31) / 32, T, 0, stream>>>(
      out_x, edge_attr, dist, batch, ei, tmlp,
      eW1T, eb1, eW2T, eb2, ln_e_g, ln_e_b, out_edge, E_);
}
